// CommitRankingModule_73916387164445
// MI455X (gfx1250) — matrix operand order repaired and checked by reading, not compiled, not run on hardware
//
#include <hip/hip_runtime.h>
#include <math.h>

#define NNODES 262144
#define CC 100
#define HID 256
#define FRAG 8
#define MPAD 112

enum { FB = 1, FGELU = 2, FRES = 4, FLN = 8, FMASK = 16 };

typedef __attribute__((ext_vector_type(8)))  float              v8f;
typedef __attribute__((ext_vector_type(16))) __bf16             v16bf;
typedef __attribute__((ext_vector_type(2)))  unsigned long long v2u64;
union BF16x16 { v16bf b; v2u64 qq[2]; unsigned long long q[4]; };

__device__ __forceinline__ unsigned short f2bf(float f) {
  unsigned b = __float_as_uint(f);
  b += 0x7FFFu + ((b >> 16) & 1u);          // round-to-nearest-even
  return (unsigned short)(b >> 16);
}
__device__ __forceinline__ float gelu_exact(float v) {
  return 0.5f * v * (1.0f + erff(v * 0.70710678118654752f));
}
// monotone float->uint key so max reduction can use integer atomics (deterministic)
__device__ __forceinline__ unsigned fkey(float f) {
  unsigned b = __float_as_uint(f);
  return (b & 0x80000000u) ? ~b : (b | 0x80000000u);
}
__device__ __forceinline__ float fkey_dec(unsigned k) {
  unsigned b = (k & 0x80000000u) ? (k & 0x7FFFFFFFu) : ~k;
  return __uint_as_float(b);
}

// ---------------------------------------------------------------- init
__global__ void kinit(unsigned* mkey, unsigned* counts) {
  int i = blockIdx.x * 256 + threadIdx.x;
  if (i < CC * 8) mkey[i] = 0u;             // key(-inf) floor
  if (i < CC) counts[i] = 0u;
}

// ---------------------------------------------------------------- fp32 -> bf16 streaming convert
__global__ void kcvt(const float* __restrict__ in, unsigned short* __restrict__ outp, int n) {
  int i = blockIdx.x * 256 + threadIdx.x;
  int stride = gridDim.x * 256;
  for (; i < n; i += stride) outp[i] = f2bf(in[i]);
}

// ---------------------------------------------------------------- fold queries into k_w
// qk_w[h,j] = scale * sum_d q[h,d] * k_w[h*32+d, j];  qk_b[h] = scale * q_h . k_b_h
__global__ void k0_prep(const float* __restrict__ q, const float* __restrict__ kw,
                        const float* __restrict__ kb, float* __restrict__ qkw,
                        float* __restrict__ qkb) {
  const float scale = 0.17677669529663687f; // 1/sqrt(32)
  int h = blockIdx.x, j = threadIdx.x;
  float s = 0.f;
  for (int d = 0; d < 32; ++d) s += q[h * 32 + d] * kw[(h * 32 + d) * HID + j];
  qkw[h * HID + j] = s * scale;
  if (j == 0) {
    float sb = 0.f;
    for (int d = 0; d < 32; ++d) sb += q[h * 32 + d] * kb[h * 32 + d];
    qkb[h] = sb * scale;
  }
}

// ---------------------------------------------------------------- pass 1: scores + segmax + counts
__global__ void k1_scores(const float* __restrict__ x, const int* __restrict__ ci,
                          const float* __restrict__ qkw, const float* __restrict__ qkb,
                          float* __restrict__ scores, unsigned* __restrict__ mkey,
                          unsigned* __restrict__ counts) {
  __shared__ float sqkw[8 * HID];
  __shared__ float sqkb[8];
  __shared__ unsigned smax[CC * 8];
  __shared__ unsigned scount[CC];
  int t = threadIdx.x;
  for (int i = t; i < 8 * HID; i += 256) sqkw[i] = qkw[i];
  if (t < 8) sqkb[t] = qkb[t];
  for (int i = t; i < CC * 8; i += 256) smax[i] = 0u;
  if (t < CC) scount[t] = 0u;
  __syncthreads();

  int w = t >> 5, lane = t & 31;
  int base = blockIdx.x * 512;
  for (int it = 0; it < 64; ++it) {
    int n = base + it * 8 + w;
    float xv[8];
#pragma unroll
    for (int i = 0; i < 8; ++i) xv[i] = x[(size_t)n * HID + lane + 32 * i];
    float p[8];
#pragma unroll
    for (int h = 0; h < 8; ++h) {
      float s = 0.f;
#pragma unroll
      for (int i = 0; i < 8; ++i) s += xv[i] * sqkw[h * HID + lane + 32 * i];
      p[h] = s;
    }
#pragma unroll
    for (int h = 0; h < 8; ++h)
#pragma unroll
      for (int off = 16; off > 0; off >>= 1) p[h] += __shfl_xor(p[h], off, 32);
    if (lane == 0) {
      int c = ci[n];
      float s8[8];
#pragma unroll
      for (int h = 0; h < 8; ++h) s8[h] = p[h] + sqkb[h];
      float4 v0; v0.x = s8[0]; v0.y = s8[1]; v0.z = s8[2]; v0.w = s8[3];
      float4 v1; v1.x = s8[4]; v1.y = s8[5]; v1.z = s8[6]; v1.w = s8[7];
      *(float4*)&scores[(size_t)n * 8] = v0;
      *(float4*)&scores[(size_t)n * 8 + 4] = v1;
#pragma unroll
      for (int h = 0; h < 8; ++h) atomicMax(&smax[c * 8 + h], fkey(s8[h]));
      atomicAdd(&scount[c], 1u);
    }
  }
  __syncthreads();
  for (int i = t; i < CC * 8; i += 256)
    if (smax[i]) atomicMax(&mkey[i], smax[i]);
  if (t < CC && scount[t]) atomicAdd(&counts[t], scount[t]);
}

// ---------------------------------------------------------------- pass 2: e-weighted sums (deterministic)
// block (c, f): scan node range of fragment f, compact nodes with commit==c in
// ascending order (ballot scan), accumulate S[c,h,:] in registers.
__global__ void k2_partial(const float* __restrict__ x, const int* __restrict__ ci,
                           const float* __restrict__ scores, const unsigned* __restrict__ mkey,
                           float* __restrict__ pS, float* __restrict__ pD) {
  __shared__ int list[2048];
  __shared__ float mrow[8];
  __shared__ int wsum[8], wbase[8];
  __shared__ int listn;
  int c = blockIdx.x, f = blockIdx.y, t = threadIdx.x;
  int w = t >> 5, lane = t & 31;
  int h = w, d0 = lane;
  if (t < 8) mrow[t] = fkey_dec(mkey[c * 8 + t]);
  float acc[8];
#pragma unroll
  for (int i = 0; i < 8; ++i) acc[i] = 0.f;
  float dacc = 0.f;
  const int range = NNODES / FRAG;
  int rbase = f * range;

  for (int tile = 0; tile < range; tile += 2048) {
    if (t == 0) listn = 0;
    __syncthreads();
    for (int s = 0; s < 8; ++s) {
      int n = rbase + tile + s * 256 + t;
      bool flag = (ci[n] == c);
      unsigned mask = __builtin_amdgcn_ballot_w32(flag);
      int rank = __popc(mask & ((1u << lane) - 1u));
      if (lane == 0) wsum[w] = __popc(mask);
      __syncthreads();
      if (t == 0) {
        int run = listn;
        for (int i = 0; i < 8; ++i) { wbase[i] = run; run += wsum[i]; }
        listn = run;
      }
      __syncthreads();
      if (flag) list[wbase[w] + rank] = n;
      __syncthreads();
    }
    int cnt = listn;
    for (int i = 0; i < cnt; ++i) {
      int n = list[i];
      float e = __expf(scores[(size_t)n * 8 + h] - mrow[h]);
      const float4* xr = (const float4*)(x + (size_t)n * HID + 8 * d0);
      float4 a0 = xr[0], a1 = xr[1];
      acc[0] += e * a0.x; acc[1] += e * a0.y; acc[2] += e * a0.z; acc[3] += e * a0.w;
      acc[4] += e * a1.x; acc[5] += e * a1.y; acc[6] += e * a1.z; acc[7] += e * a1.w;
      if (d0 == 0) dacc += e;
    }
    __syncthreads();
  }
  float* dst = pS + (size_t)f * (CC * 8 * HID) + (size_t)(c * 8 + h) * HID + 8 * d0;
  float4 o0; o0.x = acc[0]; o0.y = acc[1]; o0.z = acc[2]; o0.w = acc[3];
  float4 o1; o1.x = acc[4]; o1.y = acc[5]; o1.z = acc[6]; o1.w = acc[7];
  *(float4*)dst = o0; *(float4*)(dst + 4) = o1;
  if (d0 == 0) pD[f * (CC * 8) + c * 8 + h] = dacc;
}

// ---------------------------------------------------------------- reduce fragments (fixed order)
__global__ void k2b_reduce(const float* __restrict__ pS, const float* __restrict__ pD,
                           float* __restrict__ S, float* __restrict__ den) {
  int i = blockIdx.x * 256 + threadIdx.x;
  if (i < CC * 8 * HID) {
    float s = 0.f;
    for (int f = 0; f < FRAG; ++f) s += pS[(size_t)f * (CC * 8 * HID) + i];
    S[i] = s;
  }
  if (i < CC * 8) {
    float s = 0.f;
    for (int f = 0; f < FRAG; ++f) s += pD[f * (CC * 8) + i];
    den[i] = s;
  }
}

// ---------------------------------------------------------------- pooled = v_w_h @ (S/den) + v_b  (bf16 out)
__global__ void k3_pool(const float* __restrict__ S, const float* __restrict__ den,
                        const float* __restrict__ vw, const float* __restrict__ vb,
                        unsigned short* __restrict__ poolbf) {
  __shared__ float srow[8 * HID];
  int c = blockIdx.x, t = threadIdx.x;
  int h = t >> 5, d = t & 31;
  for (int i = t; i < 8 * HID; i += 256) srow[i] = S[(size_t)c * 8 * HID + i];
  __syncthreads();
  float a = 0.f;
  const float* wr = vw + (size_t)(h * 32 + d) * HID;
  for (int j = 0; j < HID; ++j) a += srow[h * HID + j] * wr[j];
  float dn = den[c * 8 + h];
  float r = (dn > 0.f) ? (a / dn + vb[h * 32 + d]) : 0.f;
  poolbf[(size_t)c * HID + h * 32 + d] = f2bf(r);
}

// ---------------------------------------------------------------- WMMA bf16 GEMM + fused epilogue
// A and W are pre-converted bf16 (raw ushort); inner loop is pure b128 loads + WMMA
// (out-of-range A rows are clamped to row M-1 so the loop is branch-free and EXEC
// stays all-1s as WMMA requires; duplicate pad rows are never stored).
// out[m, n] = epilogue( sum_k A[m,k] * W[n,k] ), 16-row x 256-col block tiles.
__global__ void gemm_bf16(const unsigned short* __restrict__ A, int lda,
                          const unsigned short* __restrict__ W, int kk,
                          const float* __restrict__ bias, float* __restrict__ out,
                          unsigned short* __restrict__ outbf, int ldo, int M, int flags,
                          const float* __restrict__ res, int ldr, const float* __restrict__ g,
                          const float* __restrict__ bvec, const unsigned* __restrict__ counts) {
  __shared__ float sT[16 * 272];
  __shared__ float redS[256], redQ[256];
  __shared__ float smu[16], srs[16];
  int t = threadIdx.x, w = t >> 5, lane = t & 31;
  int r = lane & 15, hh = lane >> 4;
  int row0 = blockIdx.x * 16;
  int colbase = blockIdx.y * 256 + w * 32;

  v8f c0 = {}, c1 = {};
  int arow = row0 + r;
  if (arow >= M) arow = M - 1;               // clamp: keeps loop uniform, pad rows unused
  const unsigned short* ap  = A + (size_t)arow * lda + hh * 16;       // A: K contiguous per lane
  const unsigned short* w0p = W + (size_t)(colbase + r) * kk + hh * 16; // B: K contiguous per lane
  const unsigned short* w1p = W + (size_t)(colbase + 16 + r) * kk + hh * 16;

#pragma unroll 4
  for (int kt = 0; kt < kk; kt += 32) {
    BF16x16 a, b0, b1;
    a.qq[0] = *(const v2u64*)(ap + kt);        // K = hh*16 + 0..7
    a.qq[1] = *(const v2u64*)(ap + kt + 8);    // K = hh*16 + 8..15
    b0.qq[0] = *(const v2u64*)(w0p + kt);
    b0.qq[1] = *(const v2u64*)(w0p + kt + 8);
    b1.qq[0] = *(const v2u64*)(w1p + kt);
    b1.qq[1] = *(const v2u64*)(w1p + kt + 8);
    c0 = __builtin_amdgcn_wmma_f32_16x16x32_bf16(false, a.b, false, b0.b, (short)0, c0, false, false);
    c1 = __builtin_amdgcn_wmma_f32_16x16x32_bf16(false, a.b, false, b1.b, (short)0, c1, false, false);
  }

#pragma unroll
  for (int rr = 0; rr < 8; ++rr) {              // D layout: row = rr + 8*hh, col = lane&15
    int orow = rr + hh * 8;
    sT[orow * 272 + w * 32 + r] = c0[rr];
    sT[orow * 272 + w * 32 + r + 16] = c1[rr];
  }
  __syncthreads();

  int erow = t >> 4, j0 = t & 15;
  int growl = row0 + erow;
  float psum = 0.f, psq = 0.f;
  for (int jj = 0; jj < 16; ++jj) {
    int col = j0 * 16 + jj;
    int gcol = blockIdx.y * 256 + col;
    float v = sT[erow * 272 + col];
    if (flags & FB) v += bias[gcol];
    if (flags & FRES) v += (growl < M) ? res[(size_t)growl * ldr + gcol] : 0.f;
    if (flags & FGELU) v = gelu_exact(v);
    if (flags & FLN) { sT[erow * 272 + col] = v; psum += v; psq += v * v; }
    else if (growl < M) {
      if (out)   out[(size_t)growl * ldo + gcol] = v;
      if (outbf) outbf[(size_t)growl * ldo + gcol] = f2bf(v);
    }
  }
  if (flags & FLN) {
    redS[t] = psum; redQ[t] = psq;
    __syncthreads();
    if (j0 == 0) {
      float s = 0.f, q = 0.f;
      for (int i = 0; i < 16; ++i) { s += redS[erow * 16 + i]; q += redQ[erow * 16 + i]; }
      float mu = s * (1.0f / 256.0f);
      float var = q * (1.0f / 256.0f) - mu * mu;
      smu[erow] = mu; srs[erow] = rsqrtf(var + 1e-5f);
    }
    __syncthreads();
    float mu = smu[erow], rs = srs[erow];
    if (growl < M) {
      for (int jj = 0; jj < 16; ++jj) {
        int col = j0 * 16 + jj;
        int gcol = blockIdx.y * 256 + col;
        float v = (sT[erow * 272 + col] - mu) * rs * g[gcol] + bvec[gcol];
        if ((flags & FMASK) && counts[growl] == 0u) v = 0.f;
        if (out)   out[(size_t)growl * ldo + gcol] = v;
        if (outbf) outbf[(size_t)growl * ldo + gcol] = f2bf(v);
      }
    }
  }
}

// ---------------------------------------------------------------- per-head attention (C=100), bf16 out
__global__ void k6_attn(const float* __restrict__ qkv, unsigned short* __restrict__ obf) {
  __shared__ float Ks[CC * 32], Vs[CC * 32];
  int h = blockIdx.x, t = threadIdx.x;
  for (int i = t; i < CC * 32; i += 128) {
    int rr = i >> 5, d = i & 31;
    Ks[i] = qkv[(size_t)rr * 768 + 256 + h * 32 + d];
    Vs[i] = qkv[(size_t)rr * 768 + 512 + h * 32 + d];
  }
  __syncthreads();
  if (t >= CC) return;
  float q[32];
#pragma unroll
  for (int d = 0; d < 32; ++d) q[d] = qkv[(size_t)t * 768 + h * 32 + d];
  const float scale = 0.17677669529663687f;
  float mrun = -1e30f, l = 0.f, oacc[32];
#pragma unroll
  for (int d = 0; d < 32; ++d) oacc[d] = 0.f;
  for (int j = 0; j < CC; ++j) {
    float s = 0.f;
#pragma unroll
    for (int d = 0; d < 32; ++d) s += q[d] * Ks[j * 32 + d];
    s *= scale;
    float mnew = fmaxf(mrun, s);
    float corr = __expf(mrun - mnew);
    float p = __expf(s - mnew);
    l = l * corr + p;
#pragma unroll
    for (int d = 0; d < 32; ++d) oacc[d] = oacc[d] * corr + p * Vs[j * 32 + d];
    mrun = mnew;
  }
  float inv = 1.f / l;
#pragma unroll
  for (int d = 0; d < 32; ++d) obf[(size_t)t * HID + h * 32 + d] = f2bf(oacc[d] * inv);
}

// ---------------------------------------------------------------- ranking head
__global__ void k10_head(const float* __restrict__ xa, const float* __restrict__ r1w,
                         const float* __restrict__ r1b, const float* __restrict__ r2w,
                         const float* __restrict__ r2b, float* __restrict__ out) {
  __shared__ float xr[HID];
  __shared__ float red[128];
  int c = blockIdx.x, t = threadIdx.x;
  for (int i = t; i < HID; i += 128) xr[i] = xa[(size_t)c * HID + i];
  __syncthreads();
  float a = 0.f;
  const float* wr = r1w + (size_t)t * HID;
  for (int j = 0; j < HID; ++j) a += xr[j] * wr[j];
  a += r1b[t];
  red[t] = gelu_exact(a) * r2w[t];
  __syncthreads();
  for (int s = 64; s > 0; s >>= 1) {
    if (t < s) red[t] += red[t + s];
    __syncthreads();
  }
  if (t == 0) out[c] = red[0] + r2b[0];
}

// ================================================================ host
extern "C" void kernel_launch(void* const* d_in, const int* in_sizes, int n_in,
                              void* d_out, int out_size, void* d_ws, size_t ws_size,
                              hipStream_t stream) {
  const float* x     = (const float*)d_in[0];
  const int*   ci    = (const int*)d_in[1];
  const float* cq    = (const float*)d_in[3];
  const float* kw    = (const float*)d_in[4];
  const float* kb    = (const float*)d_in[5];
  const float* vw    = (const float*)d_in[6];
  const float* vb    = (const float*)d_in[7];
  const float* pow_  = (const float*)d_in[8];
  const float* pob   = (const float*)d_in[9];
  const float* png   = (const float*)d_in[10];
  const float* pnb   = (const float*)d_in[11];
  const float* tinw  = (const float*)d_in[12];
  const float* tinb  = (const float*)d_in[13];
  const float* toutw = (const float*)d_in[14];
  const float* toutb = (const float*)d_in[15];
  const float* tln1g = (const float*)d_in[16];
  const float* tln1b = (const float*)d_in[17];
  const float* tff1w = (const float*)d_in[18];
  const float* tff1b = (const float*)d_in[19];
  const float* tff2w = (const float*)d_in[20];
  const float* tff2b = (const float*)d_in[21];
  const float* tln2g = (const float*)d_in[22];
  const float* tln2b = (const float*)d_in[23];
  const float* r1w   = (const float*)d_in[24];
  const float* r1b   = (const float*)d_in[25];
  const float* r2w   = (const float*)d_in[26];
  const float* r2b   = (const float*)d_in[27];
  float* out = (float*)d_out;

  float* ws = (float*)d_ws;
  size_t o = 0;
  float* scores = ws + o; o += (size_t)NNODES * 8;
  float* pS     = ws + o; o += (size_t)FRAG * CC * 8 * HID;
  float* pD     = ws + o; o += FRAG * CC * 8;
  float* Sred   = ws + o; o += CC * 8 * HID;
  float* den    = ws + o; o += CC * 8;
  float* qkw    = ws + o; o += 8 * HID;
  float* qkb    = ws + o; o += 8;
  float* xa     = ws + o; o += MPAD * HID;
  float* xb     = ws + o; o += MPAD * HID;
  float* qkvb   = ws + o; o += MPAD * 768;
  unsigned* mkey   = (unsigned*)(ws + o); o += CC * 8;
  unsigned* counts = (unsigned*)(ws + o); o += CC;
  // bf16 (ushort) buffers -- sizes counted in float slots (2 ushort per slot)
  unsigned short* poolbf = (unsigned short*)(ws + o); o += MPAD * HID / 2;
  unsigned short* xabf   = (unsigned short*)(ws + o); o += MPAD * HID / 2;
  unsigned short* xbbf   = (unsigned short*)(ws + o); o += MPAD * HID / 2;
  unsigned short* attnbf = (unsigned short*)(ws + o); o += MPAD * HID / 2;
  unsigned short* ffbbf  = (unsigned short*)(ws + o); o += MPAD * 1024 / 2;
  unsigned short* powbf  = (unsigned short*)(ws + o); o += HID * HID / 2;
  unsigned short* tinbf  = (unsigned short*)(ws + o); o += 2 * 768 * HID / 2;
  unsigned short* toutbf = (unsigned short*)(ws + o); o += 2 * HID * HID / 2;
  unsigned short* tff1bf = (unsigned short*)(ws + o); o += 2 * 1024 * HID / 2;
  unsigned short* tff2bf = (unsigned short*)(ws + o); o += 2 * HID * 1024 / 2;

  kinit<<<4, 256, 0, stream>>>(mkey, counts);
  // pre-convert weights to bf16 once per call (streamed; ~6.6 MB fp32 in, ~3.3 MB bf16 out)
  kcvt<<<256, 256, 0, stream>>>(pow_,  powbf,  HID * HID);
  kcvt<<<512, 256, 0, stream>>>(tinw,  tinbf,  2 * 768 * HID);
  kcvt<<<256, 256, 0, stream>>>(toutw, toutbf, 2 * HID * HID);
  kcvt<<<512, 256, 0, stream>>>(tff1w, tff1bf, 2 * 1024 * HID);
  kcvt<<<512, 256, 0, stream>>>(tff2w, tff2bf, 2 * HID * 1024);

  k0_prep<<<8, 256, 0, stream>>>(cq, kw, kb, qkw, qkb);
  k1_scores<<<NNODES / 512, 256, 0, stream>>>(x, ci, qkw, qkb, scores, mkey, counts);
  k2_partial<<<dim3(CC, FRAG), 256, 0, stream>>>(x, ci, scores, mkey, pS, pD);
  k2b_reduce<<<(CC * 8 * HID + 255) / 256, 256, 0, stream>>>(pS, pD, Sred, den);
  k3_pool<<<CC, 256, 0, stream>>>(Sred, den, vw, vb, poolbf);

  gemm_bf16<<<dim3(7, 1), 256, 0, stream>>>(poolbf, HID, powbf, HID, pob, xa, xabf, HID, CC,
      FB | FLN | FMASK, nullptr, 0, png, pnb, counts);

  for (int l = 0; l < 2; ++l) {
    gemm_bf16<<<dim3(7, 3), 256, 0, stream>>>(xabf, HID, tinbf + (size_t)l * 768 * HID, HID,
        tinb + l * 768, qkvb, nullptr, 768, CC, FB, nullptr, 0, nullptr, nullptr, nullptr);
    k6_attn<<<8, 128, 0, stream>>>(qkvb, attnbf);
    gemm_bf16<<<dim3(7, 1), 256, 0, stream>>>(attnbf, HID, toutbf + (size_t)l * HID * HID, HID,
        toutb + l * HID, xb, xbbf, HID, CC, FB | FRES | FLN, xa, HID, tln1g + l * HID,
        tln1b + l * HID, nullptr);
    gemm_bf16<<<dim3(7, 4), 256, 0, stream>>>(xbbf, HID, tff1bf + (size_t)l * 1024 * HID, HID,
        tff1b + l * 1024, nullptr, ffbbf, 1024, CC, FB | FGELU, nullptr, 0, nullptr, nullptr,
        nullptr);
    gemm_bf16<<<dim3(7, 1), 256, 0, stream>>>(ffbbf, 1024, tff2bf + (size_t)l * HID * 1024, 1024,
        tff2b + l * HID, xa, xabf, HID, CC, FB | FRES | FLN, xb, HID, tln2g + l * HID,
        tln2b + l * HID, nullptr);
  }
  k10_head<<<CC, 128, 0, stream>>>(xa, r1w, r1b, r2w, r2b, out);
}
